// DeepseekV3MiniMoEMLP_59493886984597
// MI455X (gfx1250) — matrix operand order repaired and checked by reading, not compiled, not run on hardware
//
#include <hip/hip_runtime.h>
#include <cstdint>

// ---------------------------------------------------------------------------
// DeepSeek-mini MoE MLP for gfx1250 (MI455X), bf16 WMMA + async-LDS path.
// ---------------------------------------------------------------------------

typedef __attribute__((ext_vector_type(16))) __bf16 v16bf;
typedef __attribute__((ext_vector_type(8)))  float  v8f;

static constexpr int D  = 1024;
static constexpr int H  = 1024;
static constexpr int E  = 8;
static constexpr int T  = 32768;

static constexpr int BM = 128;   // token tile per workgroup
static constexpr int BN = 64;    // N block per pass
static constexpr int KC = 64;    // K chunk staged in LDS
static constexpr int KS = KC + 8;   // padded LDS stride (halves) for staging
static constexpr int HS = H + 8;    // padded LDS stride (halves) for h

// LDS: h 264192 B + A 18432 B + B 18432 B + comb 512 B = 301568 B < 320 KB/WGP
static constexpr unsigned SMEM_BYTES =
    (unsigned)((BM * HS + BM * KS + 2 * BN * KS) * 2 + BM * 4);

__device__ __forceinline__ uint16_t f32_to_bf16(float f) {
  uint32_t u = __float_as_uint(f);
  u += 0x7FFFu + ((u >> 16) & 1u);   // round-to-nearest-even
  return (uint16_t)(u >> 16);
}

// --- CDNA5 async global->LDS copy (16 B per lane), tracked by ASYNCcnt ------
__device__ __forceinline__ void async_copy16(void* lds, const void* gsrc) {
  uint32_t loff = (uint32_t)(uintptr_t)lds;   // LDS byte offset (low 32 bits)
  asm volatile("global_load_async_to_lds_b128 %0, %1, off"
               :: "v"(loff), "v"(gsrc)
               : "memory");
}
__device__ __forceinline__ void wait_async0() {
  asm volatile("s_wait_asynccnt 0" ::: "memory");
}

// A fragment (bf16 16x32): lane<16 -> K {0..7,16..23}, lane>=16 -> +8.
__device__ __forceinline__ v16bf load_a_frag(const uint16_t* base, int stride,
                                             int m, int kbase) {
  union { v16bf v; uint4 q[2]; } f;
  const uint16_t* p = base + m * stride + kbase;
  f.q[0] = *(const uint4*)(p);
  f.q[1] = *(const uint4*)(p + 16);
  return f.v;
}

// B fragment (bf16 32x16 from row-major W[n][k]): lane<16 -> K {0..7,16..23}, lane>=16 -> +8.
__device__ __forceinline__ v16bf load_b_frag(const uint16_t* base, int stride,
                                             int n, int k) {
  union { v16bf v; uint4 q[2]; } f;
  const uint16_t* p = base + n * stride + k;
  f.q[0] = *(const uint4*)(p);
  f.q[1] = *(const uint4*)(p + 16);
  return f.v;
}

__device__ __forceinline__ v8f wmma_bf16(v16bf a, v16bf b, v8f c) {
  return __builtin_amdgcn_wmma_f32_16x16x32_bf16(false, a, false, b,
                                                 (short)0, c, false, false);
}

// ---------------------------------------------------------------------------
// Kernel 1: f32 -> bf16 bulk convert (weights)
// ---------------------------------------------------------------------------
__global__ void cvt_f32_bf16_kernel(const float* __restrict__ src,
                                    uint16_t* __restrict__ dst, int n) {
  int stride = gridDim.x * blockDim.x;
  for (int i = blockIdx.x * blockDim.x + threadIdx.x; i < n; i += stride)
    dst[i] = f32_to_bf16(src[i]);
}

// ---------------------------------------------------------------------------
// Kernel 2: router (one wave32 per token) + x f32->bf16 conversion
// ---------------------------------------------------------------------------
__global__ __launch_bounds__(256) void router_kernel(
    const float* __restrict__ x, const float* __restrict__ gate_w,
    uint16_t* __restrict__ xbf, float* __restrict__ comb) {
  int token = blockIdx.x * 8 + (threadIdx.x >> 5);
  int lane  = threadIdx.x & 31;
  const float* xr = x + (size_t)token * D;

  float acc[E];
#pragma unroll
  for (int e = 0; e < E; ++e) acc[e] = 0.f;

  for (int d = lane; d < D; d += 32) {
    float xv = xr[d];
    xbf[(size_t)token * D + d] = f32_to_bf16(xv);
#pragma unroll
    for (int e = 0; e < E; ++e) acc[e] += xv * gate_w[e * D + d];
  }
#pragma unroll
  for (int e = 0; e < E; ++e) {
#pragma unroll
    for (int off = 16; off > 0; off >>= 1)
      acc[e] += __shfl_xor(acc[e], off, 32);
  }
  float s[E];
#pragma unroll
  for (int e = 0; e < E; ++e)
    s[e] = __builtin_amdgcn_rcpf(1.f + __expf(-acc[e]));

  int i1 = 0; float v1 = s[0];
#pragma unroll
  for (int e = 1; e < E; ++e) if (s[e] > v1) { v1 = s[e]; i1 = e; }
  int i2 = -1; float v2 = -1.f;
#pragma unroll
  for (int e = 0; e < E; ++e)
    if (e != i1 && s[e] > v2) { v2 = s[e]; i2 = e; }
  float inv = __builtin_amdgcn_rcpf(v1 + v2 + 1e-9f);
  if (lane < E) {
    float c = (lane == i1) ? v1 * inv : ((lane == i2) ? v2 * inv : 0.f);
    comb[(size_t)token * E + lane] = c;
  }
}

// ---------------------------------------------------------------------------
// Kernel 3: fused MoE expert GEMMs, bf16 WMMA, async staging, h in LDS.
// ---------------------------------------------------------------------------
__global__ __launch_bounds__(256) void moe_kernel(
    const uint16_t* __restrict__ xbf, const uint16_t* __restrict__ gbf,
    const uint16_t* __restrict__ ubf, const uint16_t* __restrict__ dbf,
    const float* __restrict__ comb, float* __restrict__ out) {
  extern __shared__ char smem[];
  uint16_t* hBuf = (uint16_t*)smem;                       // BM x HS
  uint16_t* aBuf = hBuf + BM * HS;                        // BM x KS
  uint16_t* bBuf = aBuf + BM * KS;                        // 2 x BN x KS
  float* combLds = (float*)(bBuf + 2 * BN * KS);          // BM

  const int tid  = threadIdx.x;
  const int lane = tid & 31;
  const int wid  = tid >> 5;
  const int ln15 = lane & 15;
  const int aKoff = (lane < 16) ? 0 : 8;
  const int bKoff = (lane < 16) ? 0 : 8;
  const int m0 = (wid & 3) * 32;
  const int n0 = (wid >> 2) * 32;
  const int t0 = blockIdx.x * BM;

  // Zero this tile's output rows (accumulated across experts below).
  for (int i = tid; i < BM * D; i += 256) out[(size_t)t0 * D + i] = 0.f;
  __threadfence();
  __syncthreads();

  // Per-thread staging coordinates.
  const int ar = tid >> 1, ac = (tid & 1) << 5;   // A: 128 rows x 64 halves
  const int br = tid >> 2, bc = (tid & 3) << 4;   // B: 64 rows x 64 halves

  for (int e = 0; e < E; ++e) {
    if (tid < BM) combLds[tid] = comb[(size_t)(t0 + tid) * E + e];
    const uint16_t* Wg = gbf + (size_t)e * H * D;
    const uint16_t* Wu = ubf + (size_t)e * H * D;
    const uint16_t* Wd = dbf + (size_t)e * D * H;

    // ---- GEMM1: h = silu(x*Wg^T) * (x*Wu^T) -> LDS hBuf ----
    for (int nb = 0; nb < H / BN; ++nb) {
      v8f cg[2][2], cu[2][2];
#pragma unroll
      for (int sm = 0; sm < 2; ++sm)
#pragma unroll
        for (int sn = 0; sn < 2; ++sn) { cg[sm][sn] = v8f{}; cu[sm][sn] = v8f{}; }

      for (int k0 = 0; k0 < D; k0 += KC) {
        __syncthreads();   // staging buffers free to overwrite
        {  // x tile: 4 x 16B async per thread
          const uint16_t* s = xbf + (size_t)(t0 + ar) * D + k0 + ac;
          uint16_t* d2 = aBuf + ar * KS + ac;
#pragma unroll
          for (int j = 0; j < 4; ++j) async_copy16(d2 + 8 * j, s + 8 * j);
        }
        {  // gate & up weights: 2 x 16B async per thread each
          const uint16_t* sg = Wg + (size_t)(nb * BN + br) * D + k0 + bc;
          const uint16_t* su = Wu + (size_t)(nb * BN + br) * D + k0 + bc;
          uint16_t* dg = bBuf + br * KS + bc;
          uint16_t* du = bBuf + BN * KS + br * KS + bc;
          async_copy16(dg, sg);     async_copy16(dg + 8, sg + 8);
          async_copy16(du, su);     async_copy16(du + 8, su + 8);
        }
        wait_async0();
        __syncthreads();
#pragma unroll
        for (int ks = 0; ks < KC; ks += 32) {
          v16bf aF[2], bG[2], bU[2];
#pragma unroll
          for (int sm = 0; sm < 2; ++sm)
            aF[sm] = load_a_frag(aBuf, KS, m0 + sm * 16 + ln15, ks + aKoff);
#pragma unroll
          for (int sn = 0; sn < 2; ++sn) {
            bG[sn] = load_b_frag(bBuf, KS, n0 + sn * 16 + ln15, ks + bKoff);
            bU[sn] = load_b_frag(bBuf + BN * KS, KS, n0 + sn * 16 + ln15, ks + bKoff);
          }
#pragma unroll
          for (int sm = 0; sm < 2; ++sm)
#pragma unroll
            for (int sn = 0; sn < 2; ++sn) {
              cg[sm][sn] = wmma_bf16(aF[sm], bG[sn], cg[sm][sn]);
              cu[sm][sn] = wmma_bf16(aF[sm], bU[sn], cu[sm][sn]);
            }
        }
      }
      // silu(g) * u -> bf16 -> hBuf (hardware rcp for the sigmoid)
#pragma unroll
      for (int sm = 0; sm < 2; ++sm)
#pragma unroll
        for (int sn = 0; sn < 2; ++sn)
#pragma unroll
          for (int r = 0; r < 8; ++r) {
            int mloc = m0 + sm * 16 + ((lane & 16) >> 1) + r;
            int ncol = nb * BN + n0 + sn * 16 + ln15;
            float g = cg[sm][sn][r], u = cu[sm][sn][r];
            float hv = g * __builtin_amdgcn_rcpf(1.f + __expf(-g)) * u;
            hBuf[mloc * HS + ncol] = f32_to_bf16(hv);
          }
    }
    __syncthreads();   // full h visible to all waves

    // ---- GEMM2: out += (h * Wd^T) * comb[:, e] ----
    for (int nb2 = 0; nb2 < D / BN; ++nb2) {
      v8f c2[2][2];
#pragma unroll
      for (int sm = 0; sm < 2; ++sm)
#pragma unroll
        for (int sn = 0; sn < 2; ++sn) c2[sm][sn] = v8f{};

      for (int k0 = 0; k0 < H; k0 += KC) {
        __syncthreads();
        {  // down weights: 2 x 16B async per thread
          const uint16_t* s = Wd + (size_t)(nb2 * BN + br) * H + k0 + bc;
          uint16_t* d2 = bBuf + br * KS + bc;
          async_copy16(d2, s);   async_copy16(d2 + 8, s + 8);
        }
        wait_async0();
        __syncthreads();
#pragma unroll
        for (int ks = 0; ks < KC; ks += 32) {
          v16bf aF[2], bD[2];
#pragma unroll
          for (int sm = 0; sm < 2; ++sm)
            aF[sm] = load_a_frag(hBuf, HS, m0 + sm * 16 + ln15, k0 + ks + aKoff);
#pragma unroll
          for (int sn = 0; sn < 2; ++sn)
            bD[sn] = load_b_frag(bBuf, KS, n0 + sn * 16 + ln15, ks + bKoff);
#pragma unroll
          for (int sm = 0; sm < 2; ++sm)
#pragma unroll
            for (int sn = 0; sn < 2; ++sn)
              c2[sm][sn] = wmma_bf16(aF[sm], bD[sn], c2[sm][sn]);
        }
      }
#pragma unroll
      for (int sm = 0; sm < 2; ++sm)
#pragma unroll
        for (int sn = 0; sn < 2; ++sn)
#pragma unroll
          for (int r = 0; r < 8; ++r) {
            int mloc = m0 + sm * 16 + ((lane & 16) >> 1) + r;
            int dcol = nb2 * BN + n0 + sn * 16 + ln15;
            size_t idx = (size_t)(t0 + mloc) * D + dcol;
            out[idx] += c2[sm][sn][r] * combLds[mloc];
          }
      __syncthreads();   // out RMW done before bBuf restage / next expert
    }
  }
}

// ---------------------------------------------------------------------------
extern "C" void kernel_launch(void* const* d_in, const int* in_sizes, int n_in,
                              void* d_out, int out_size, void* d_ws, size_t ws_size,
                              hipStream_t stream) {
  const float* x       = (const float*)d_in[0];
  const float* gate_w  = (const float*)d_in[1];
  const float* gate_pj = (const float*)d_in[2];
  const float* up_pj   = (const float*)d_in[3];
  const float* down_pj = (const float*)d_in[4];
  float* out = (float*)d_out;

  char* w = (char*)d_ws;
  uint16_t* xbf = (uint16_t*)w;                 w += (size_t)T * D * 2;
  uint16_t* gbf = (uint16_t*)w;                 w += (size_t)E * H * D * 2;
  uint16_t* ubf = (uint16_t*)w;                 w += (size_t)E * H * D * 2;
  uint16_t* dbf = (uint16_t*)w;                 w += (size_t)E * D * H * 2;
  float*   comb = (float*)w;

  const int nW = E * H * D;
  cvt_f32_bf16_kernel<<<4096, 256, 0, stream>>>(gate_pj, gbf, nW);
  cvt_f32_bf16_kernel<<<4096, 256, 0, stream>>>(up_pj,   ubf, nW);
  cvt_f32_bf16_kernel<<<4096, 256, 0, stream>>>(down_pj, dbf, nW);

  router_kernel<<<T / 8, 256, 0, stream>>>(x, gate_w, xbf, comb);

  moe_kernel<<<T / BM, 256, SMEM_BYTES, stream>>>(xbf, gbf, ubf, dbf, comb, out);
}
